// DelayedSelfAttention_7954279432250
// MI455X (gfx1250) — matrix operand order repaired and checked by reading, not compiled, not run on hardware
//
#include <hip/hip_runtime.h>

// ---------------------------------------------------------------------------
// Problem constants (B=4, T=1024, C=1024, H=16, head=64, seq=2T=2048)
// ---------------------------------------------------------------------------
#define BATCH 4
#define T1    1024
#define S2    2048
#define CDIM  1024
#define N3C   3072
#define NH    16
#define HD    64
#define LORA_SCALE 2.0f      // ALPHA / RANK = 16 / 8
#define NEG_BIG (-1e30f)

typedef __attribute__((ext_vector_type(16))) __bf16 v16bf;
typedef __attribute__((ext_vector_type(8)))  __bf16 v8bf;
typedef __attribute__((ext_vector_type(8)))  float  v8f;

__device__ __forceinline__ v8f zero8() {
  v8f v;
#pragma unroll
  for (int i = 0; i < 8; ++i) v[i] = 0.0f;
  return v;
}

// A-operand fragment (16x32 bf16, interleaved K): p[base..base+7], p[base+16..base+23]
__device__ __forceinline__ v16bf load_frag_a(const __bf16* p, int base) {
  v8bf lo = *(const v8bf*)(p + base);
  v8bf hi = *(const v8bf*)(p + base + 16);
  v16bf r;
#pragma unroll
  for (int i = 0; i < 8; ++i) { r[i] = lo[i]; r[i + 8] = hi[i]; }
  return r;
}

// B-operand fragment (32x16 bf16, interleaved K): p[0..7], p[16..23]
__device__ __forceinline__ v16bf load_frag_b(const __bf16* p) {
  v8bf lo = *(const v8bf*)(p);
  v8bf hi = *(const v8bf*)(p + 16);
  v16bf r;
#pragma unroll
  for (int i = 0; i < 8; ++i) { r[i] = lo[i]; r[i + 8] = hi[i]; }
  return r;
}

__device__ __forceinline__ v8f wmma_bf(v16bf a, v16bf b, v8f c) {
  return __builtin_amdgcn_wmma_f32_16x16x32_bf16(false, a, false, b, (short)0, c,
                                                 false, false);
}

// reductions within each 16-lane half of a wave32
__device__ __forceinline__ float rmax16(float v) {
  v = fmaxf(v, __shfl_xor(v, 1, 32));
  v = fmaxf(v, __shfl_xor(v, 2, 32));
  v = fmaxf(v, __shfl_xor(v, 4, 32));
  v = fmaxf(v, __shfl_xor(v, 8, 32));
  return v;
}
__device__ __forceinline__ float rsum16(float v) {
  v += __shfl_xor(v, 1, 32);
  v += __shfl_xor(v, 2, 32);
  v += __shfl_xor(v, 4, 32);
  v += __shfl_xor(v, 8, 32);
  return v;
}

// delayed-attention mask (LOOKAHEAD=64, OVERLAP=64)
__device__ __forceinline__ bool mask_ok(int q, int k) {
  if (q < T1) {
    if (k < T1) return (k <= q) && (k >= q - 128);        // e1->e1 band
    return (k - T1) <= (q - 64);                          // e1->e2 delayed causal
  } else {
    if (k < T1) return (k >= q - T1 - 64) && (k <= q - T1 + 64); // e2->e1 band
    return k <= q;                                        // e2->e2 causal
  }
}

// ---------------------------------------------------------------------------
// K0: casts + weight transposes (Wt stored [N][K] so B-fragments are contiguous)
// ---------------------------------------------------------------------------
__global__ void cast_prep_kernel(const float* __restrict__ e1, const float* __restrict__ e2,
                                 const float* __restrict__ Wa, const float* __restrict__ Wp,
                                 __bf16* __restrict__ Xb, __bf16* __restrict__ Wat,
                                 __bf16* __restrict__ Wpt) {
  const size_t idx = (size_t)blockIdx.x * blockDim.x + threadIdx.x;
  const size_t n1 = (size_t)BATCH * S2 * CDIM;   // 8388608
  const size_t n2 = (size_t)N3C * CDIM;          // 3145728
  if (idx < n1) {
    const size_t mrow = idx >> 10;
    const int c  = (int)(idx & 1023);
    const int b  = (int)(mrow >> 11);
    const int tt = (int)(mrow & 2047);
    const float v = (tt < T1) ? e1[((size_t)b * T1 + tt) * CDIM + c]
                              : e2[((size_t)b * T1 + (tt - T1)) * CDIM + c];
    Xb[idx] = (__bf16)v;
  } else if (idx < n1 + n2) {
    const size_t i = idx - n1;
    const size_t nrow = i >> 10;
    const int k = (int)(i & 1023);
    Wat[i] = (__bf16)Wa[(size_t)k * N3C + nrow];
  } else {
    const size_t i = idx - n1 - n2;
    const size_t nrow = i >> 10;
    const int k = (int)(i & 1023);
    Wpt[i] = (__bf16)Wp[(size_t)k * CDIM + nrow];
  }
}

// ---------------------------------------------------------------------------
// rank-8 LoRA left factors (tiny fp32 work)
// ---------------------------------------------------------------------------
__global__ void lora_in_kernel(const float* __restrict__ e2, const float* __restrict__ A,
                               float* __restrict__ out) {
  const int row = blockIdx.x * blockDim.x + threadIdx.x;   // 0..4095 (b*T+tt)
  const float* x = e2 + (size_t)row * CDIM;
  float acc[8];
#pragma unroll
  for (int r = 0; r < 8; ++r) acc[r] = 0.0f;
  for (int k = 0; k < CDIM; ++k) {
    const float xv = x[k];
    const float* ar = A + (size_t)k * 8;
#pragma unroll
    for (int r = 0; r < 8; ++r) acc[r] += xv * ar[r];
  }
#pragma unroll
  for (int r = 0; r < 8; ++r) out[(size_t)row * 8 + r] = acc[r];
}

__global__ void lora_y_kernel(const __bf16* __restrict__ Yb, const float* __restrict__ A,
                              float* __restrict__ out) {
  const int row = blockIdx.x * blockDim.x + threadIdx.x;   // 0..4095 (b*T+tt)
  const int b = row >> 10, tt = row & 1023;
  const __bf16* x = Yb + ((size_t)b * S2 + T1 + tt) * CDIM;
  float acc[8];
#pragma unroll
  for (int r = 0; r < 8; ++r) acc[r] = 0.0f;
  for (int k = 0; k < CDIM; ++k) {
    const float xv = (float)x[k];
    const float* ar = A + (size_t)k * 8;
#pragma unroll
    for (int r = 0; r < 8; ++r) acc[r] += xv * ar[r];
  }
#pragma unroll
  for (int r = 0; r < 8; ++r) out[(size_t)row * 8 + r] = acc[r];
}

// ---------------------------------------------------------------------------
// WMMA GEMM core: one wave -> 16 rows x 64 cols, K-loop in steps of 32
// ---------------------------------------------------------------------------
__device__ __forceinline__ void gemm_core(const __bf16* __restrict__ A,
                                          const __bf16* __restrict__ Bt, int Kdim,
                                          int m0, int n0, int lane, v8f acc[4]) {
  const int l16 = lane & 15;
  const int abase = (lane >> 4) * 8;
  const int bbase = (lane >> 4) * 8;
  const __bf16* arow = A  + (size_t)(m0 + l16) * Kdim;
  const __bf16* brow = Bt + (size_t)(n0 + l16) * Kdim;
  for (int k0 = 0; k0 < Kdim; k0 += 32) {
    const v16bf af = load_frag_a(arow, k0 + abase);
#pragma unroll
    for (int tn = 0; tn < 4; ++tn) {
      const v16bf bf = load_frag_b(brow + (size_t)tn * 16 * Kdim + k0 + bbase);
      acc[tn] = wmma_bf(af, bf, acc[tn]);
    }
  }
}

// ---------------------------------------------------------------------------
// K2: QKV = X @ W_attn^T(t) (+LoRA on e2 rows), scatter to Q,K,[Vt] bf16 layouts
// ---------------------------------------------------------------------------
__global__ void qkv_gemm_kernel(const __bf16* __restrict__ Xb, const __bf16* __restrict__ Wat,
                                const float* __restrict__ lt1, const float* __restrict__ lBa,
                                __bf16* __restrict__ Qb, __bf16* __restrict__ Kb,
                                __bf16* __restrict__ Vtb) {
  const int lane = threadIdx.x & 31;
  const int wave = threadIdx.x >> 5;
  const int m0 = blockIdx.y * 64 + wave * 16;
  const int n0 = blockIdx.x * 64;
  v8f acc[4];
#pragma unroll
  for (int t = 0; t < 4; ++t) acc[t] = zero8();
  gemm_core(Xb, Wat, CDIM, m0, n0, lane, acc);

  const int l16 = lane & 15, half = lane >> 4;
#pragma unroll
  for (int tn = 0; tn < 4; ++tn) {
    const int n = n0 + tn * 16 + l16;
    float lb[8];
#pragma unroll
    for (int i = 0; i < 8; ++i) lb[i] = lBa[(size_t)i * N3C + n];
    const int seg = n >> 10;            // 0=Q 1=K 2=V
    const int nn = n & 1023;
    const int h = nn >> 6, d = nn & 63;
#pragma unroll
    for (int j = 0; j < 8; ++j) {
      const int r  = m0 + j + half * 8;
      const int b  = r >> 11;
      const int tt = r & 2047;
      float v = acc[tn][j];
      if (tt >= T1) {                   // LoRA only on the e2 half
        const float* lt = lt1 + (size_t)(b * T1 + tt - T1) * 8;
        float dsum = 0.0f;
#pragma unroll
        for (int i = 0; i < 8; ++i) dsum += lt[i] * lb[i];
        v += LORA_SCALE * dsum;
      }
      const __bf16 bv = (__bf16)v;
      const size_t bh = (size_t)b * NH + h;
      if (seg == 0)      Qb [(bh * S2 + tt) * HD + d] = bv;
      else if (seg == 1) Kb [(bh * S2 + tt) * HD + d] = bv;
      else               Vtb[(bh * HD + d) * S2 + tt] = bv;   // V transposed
    }
  }
}

// ---------------------------------------------------------------------------
// K3: flash attention, one wave = 16 queries, key blocks of 32, online softmax
// ---------------------------------------------------------------------------
__global__ void attn_kernel(const __bf16* __restrict__ Qb, const __bf16* __restrict__ Kb,
                            const __bf16* __restrict__ Vtb, __bf16* __restrict__ Yb) {
  __shared__ __bf16 plds[4][16 * 32];
  const int lane = threadIdx.x & 31;
  const int wave = threadIdx.x >> 5;
  const int l16 = lane & 15;
  const int half = lane >> 4;
  const int abase = half * 8;
  const int bbase = half * 8;
  const int bh = blockIdx.y;
  const int b = bh >> 4, h = bh & 15;
  const int q0 = blockIdx.x * 64 + wave * 16;
  const __bf16* Q  = Qb  + (size_t)bh * S2 * HD;
  const __bf16* K  = Kb  + (size_t)bh * S2 * HD;
  const __bf16* Vt = Vtb + (size_t)bh * HD * S2;
  __bf16* lds = plds[wave];

  v16bf qf[2];
  {
    const __bf16* qp = Q + (size_t)(q0 + l16) * HD;
    qf[0] = load_frag_a(qp, 0 + abase);
    qf[1] = load_frag_a(qp, 32 + abase);
  }
  v8f o[4];
  float m[8], l[8];
#pragma unroll
  for (int t = 0; t < 4; ++t) o[t] = zero8();
#pragma unroll
  for (int j = 0; j < 8; ++j) { m[j] = NEG_BIG; l[j] = 0.0f; }

  // wave-uniform allowed key intervals for this 16-query block
  int a1lo, a1hi, a2hi;
  if (q0 < T1) { a1lo = q0 - 128;      a1hi = q0 + 15;             a2hi = T1 + q0 + 15 - 64; }
  else         { a1lo = q0 - T1 - 64;  a1hi = q0 + 15 - T1 + 64;   a2hi = q0 + 15; }

  for (int kb = 0; kb < S2; kb += 32) {
    const bool active = (kb < T1) ? ((kb <= a1hi) && (kb + 31 >= a1lo)) : (kb <= a2hi);
    if (!active) continue;

    // S = Q (16x64) @ K^T (64x32) -> two 16x16 f32 tiles
    v8f s0 = zero8(), s1 = zero8();
#pragma unroll
    for (int kc = 0; kc < 2; ++kc) {
      const v16bf b0 = load_frag_b(K + (size_t)(kb + l16) * HD      + kc * 32 + bbase);
      const v16bf b1 = load_frag_b(K + (size_t)(kb + 16 + l16) * HD + kc * 32 + bbase);
      s0 = wmma_bf(qf[kc], b0, s0);
      s1 = wmma_bf(qf[kc], b1, s1);
    }

    // mask + scale + online softmax; write P tile to LDS in A-layout order
#pragma unroll
    for (int j = 0; j < 8; ++j) {
      const int r = q0 + j + half * 8;
      const float sa = mask_ok(r, kb + l16)      ? s0[j] * 0.125f : NEG_BIG;
      const float sb = mask_ok(r, kb + 16 + l16) ? s1[j] * 0.125f : NEG_BIG;
      const float mn = fmaxf(m[j], rmax16(fmaxf(sa, sb)));
      const float alpha = __expf(m[j] - mn);
      m[j] = mn;
      const float p0 = (sa < -1e20f) ? 0.0f : __expf(sa - mn);
      const float p1 = (sb < -1e20f) ? 0.0f : __expf(sb - mn);
      l[j] = l[j] * alpha + rsum16(p0 + p1);
#pragma unroll
      for (int t = 0; t < 4; ++t) o[t][j] *= alpha;
      lds[(j + half * 8) * 32 + l16]      = (__bf16)p0;
      lds[(j + half * 8) * 32 + 16 + l16] = (__bf16)p1;
    }
    // DS ops are in-order within the wave; make data visible before reload
    asm volatile("s_wait_dscnt 0x0" ::: "memory");

    // O += P (16x32) @ V (32x64)
    const v16bf pf = load_frag_a(lds + l16 * 32, abase);
#pragma unroll
    for (int t = 0; t < 4; ++t) {
      const int d = t * 16 + l16;
      const v16bf vf = load_frag_b(Vt + (size_t)d * S2 + kb + bbase);
      o[t] = wmma_bf(pf, vf, o[t]);
    }
  }

  // normalize and write y (bf16, [B, 2T, C] with c = h*64+d)
#pragma unroll
  for (int j = 0; j < 8; ++j) {
    const int r = q0 + j + half * 8;
    const float inv = 1.0f / l[j];
#pragma unroll
    for (int t = 0; t < 4; ++t) {
      const int c = h * HD + t * 16 + l16;
      Yb[((size_t)b * S2 + r) * CDIM + c] = (__bf16)(o[t][j] * inv);
    }
  }
}

// ---------------------------------------------------------------------------
// K5: y @ W_proj (+LoRA on e2 rows), fp32 out with half-split reordering
// ---------------------------------------------------------------------------
__global__ void proj_gemm_kernel(const __bf16* __restrict__ Yb, const __bf16* __restrict__ Wpt,
                                 const float* __restrict__ lt2, const float* __restrict__ lBp,
                                 float* __restrict__ out) {
  const int lane = threadIdx.x & 31;
  const int wave = threadIdx.x >> 5;
  const int m0 = blockIdx.y * 64 + wave * 16;
  const int n0 = blockIdx.x * 64;
  v8f acc[4];
#pragma unroll
  for (int t = 0; t < 4; ++t) acc[t] = zero8();
  gemm_core(Yb, Wpt, CDIM, m0, n0, lane, acc);

  const int l16 = lane & 15, half = lane >> 4;
#pragma unroll
  for (int tn = 0; tn < 4; ++tn) {
    const int n = n0 + tn * 16 + l16;
    float lb[8];
#pragma unroll
    for (int i = 0; i < 8; ++i) lb[i] = lBp[(size_t)i * CDIM + n];
#pragma unroll
    for (int j = 0; j < 8; ++j) {
      const int r  = m0 + j + half * 8;
      const int b  = r >> 11;
      const int tt = r & 2047;
      float v = acc[tn][j];
      if (tt >= T1) {
        const float* lt = lt2 + (size_t)(b * T1 + tt - T1) * 8;
        float dsum = 0.0f;
#pragma unroll
        for (int i = 0; i < 8; ++i) dsum += lt[i] * lb[i];
        v += LORA_SCALE * dsum;
      }
      const size_t oidx = (tt < T1)
          ? (((size_t)b * T1 + tt) * CDIM + n)
          : ((size_t)BATCH * T1 * CDIM + ((size_t)b * T1 + (tt - T1)) * CDIM + n);
      out[oidx] = v;
    }
  }
}

// ---------------------------------------------------------------------------
// Host launch
// ---------------------------------------------------------------------------
extern "C" void kernel_launch(void* const* d_in, const int* in_sizes, int n_in,
                              void* d_out, int out_size, void* d_ws, size_t ws_size,
                              hipStream_t stream) {
  (void)in_sizes; (void)n_in; (void)out_size; (void)ws_size;
  const float* e1  = (const float*)d_in[0];
  const float* e2  = (const float*)d_in[1];
  const float* Wa  = (const float*)d_in[2];
  const float* Wp  = (const float*)d_in[3];
  const float* lAa = (const float*)d_in[4];
  const float* lBa = (const float*)d_in[5];
  const float* lAp = (const float*)d_in[6];
  const float* lBp = (const float*)d_in[7];

  char* ws = (char*)d_ws;
  __bf16* Xb  = (__bf16*)(ws + 0);          // 16 MB  [8192][1024] bf16
  __bf16* Wat = (__bf16*)(ws + 16777216);   //  6 MB  [3072][1024] bf16 (W_attn^T)
  __bf16* Wpt = (__bf16*)(ws + 23068672);   //  2 MB  [1024][1024] bf16 (W_proj^T)
  __bf16* Qb  = (__bf16*)(ws + 25165824);   // 16 MB  [B*H][2T][64]
  __bf16* Kb  = (__bf16*)(ws + 41943040);   // 16 MB  [B*H][2T][64]
  __bf16* Vtb = (__bf16*)(ws + 58720256);   // 16 MB  [B*H][64][2T]
  __bf16* Yb  = (__bf16*)(ws + 75497472);   // 16 MB  [B][2T][C]
  float*  l1  = (float*) (ws + 92274688);   // 128 KB [B*T][8]
  float*  l2  = (float*) (ws + 92405760);   // 128 KB [B*T][8]

  cast_prep_kernel<<<49152, 256, 0, stream>>>(e1, e2, Wa, Wp, Xb, Wat, Wpt);
  lora_in_kernel<<<16, 256, 0, stream>>>(e2, lAa, l1);
  qkv_gemm_kernel<<<dim3(48, 128), 128, 0, stream>>>(Xb, Wat, l1, lBa, Qb, Kb, Vtb);
  attn_kernel<<<dim3(32, 64), 128, 0, stream>>>(Qb, Kb, Vtb, Yb);
  lora_y_kernel<<<16, 256, 0, stream>>>(Yb, lAp, l2);
  proj_gemm_kernel<<<dim3(16, 128), 128, 0, stream>>>(Yb, Wpt, l2, lBp, (float*)d_out);
}
